// SpotlightAttention_65524021067966
// MI455X (gfx1250) — matrix operand order repaired and checked by reading, not compiled, not run on hardware
//
#include <hip/hip_runtime.h>

// ---------------- types & helpers ----------------
typedef __bf16 bf_t;
typedef __attribute__((ext_vector_type(16))) __bf16 bf16x16;
typedef __attribute__((ext_vector_type(8)))  __bf16 bf16x8;
typedef __attribute__((ext_vector_type(8)))  float  f32x8;

#define Bb   16
#define Cc   256
#define Mm   4096
#define HID  256
#define Nn   300
#define NPAD 304
#define SDd  64
#define NHh  4

__device__ __forceinline__ float bf2f(bf_t b) {
  union { unsigned short s; bf_t b; } i; i.b = b;
  union { unsigned u; float f; } o; o.u = ((unsigned)i.s) << 16; return o.f;
}
__device__ __forceinline__ bf_t f2bf(float f) {
  union { float f; unsigned u; } a; a.f = f;
  unsigned r = a.u + 0x7FFFu + ((a.u >> 16) & 1u);
  union { unsigned short s; bf_t b; } o; o.s = (unsigned short)(r >> 16); return o.b;
}
__device__ __forceinline__ f32x8 zero8() {
  f32x8 z; 
#pragma unroll
  for (int i = 0; i < 8; ++i) z[i] = 0.f; 
  return z;
}
__device__ __forceinline__ bf16x16 bzero16() {
  bf16x16 z;
#pragma unroll
  for (int i = 0; i < 16; ++i) z[i] = f2bf(0.f);
  return z;
}
// A fragment: lane&15 = row, k = hf*16 + e: two 8-elem chunks
__device__ __forceinline__ bf16x16 frag_a(const bf_t* row_ptr, int k0, int hf) {
  bf16x16 a;
  *(bf16x8*)&a         = *(const bf16x8*)(row_ptr + k0 + hf * 16);
  *(((bf16x8*)&a) + 1) = *(const bf16x8*)(row_ptr + k0 + hf * 16 + 8);
  return a;
}
// B fragment: lane&15 = col, k = hf*16 + e: one 16-elem chunk
__device__ __forceinline__ bf16x16 frag_b(const bf_t* col_ptr, int k0, int hf) {
  return *(const bf16x16*)(col_ptr + k0 + hf * 16);
}
__device__ __forceinline__ f32x8 wmma_bf(bf16x16 a, bf16x16 b, f32x8 c) {
  return __builtin_amdgcn_wmma_f32_16x16x32_bf16(false, a, false, b, (short)0, c, false, false);
}
__device__ __forceinline__ float redmax16(float v) {
#pragma unroll
  for (int m = 1; m < 16; m <<= 1) v = fmaxf(v, __shfl_xor(v, m, 32));
  return v;
}
__device__ __forceinline__ float redsum16(float v) {
#pragma unroll
  for (int m = 1; m < 16; m <<= 1) v += __shfl_xor(v, m, 32);
  return v;
}

// ---------------- prep kernels ----------------
__global__ void k_cast(const float* __restrict__ s, bf_t* __restrict__ d, int n) {
  for (int i = blockIdx.x * blockDim.x + threadIdx.x; i < n; i += gridDim.x * blockDim.x)
    d[i] = f2bf(s[i]);
}
__global__ void k_qpad(const float* __restrict__ q, bf_t* __restrict__ d) {
  int i = blockIdx.x * blockDim.x + threadIdx.x;
  if (i < NPAD * SDd) { int n = i / SDd; d[i] = (n < Nn) ? f2bf(q[i]) : f2bf(0.f); }
}
__global__ void k_bnfold(const float* g, const float* bb, const float* mm, const float* vv,
                         const float* fpb, float* s, float* t, float* divacc) {
  int i = threadIdx.x;
  if (i == 0) *divacc = 0.f;
  if (i < 128) {
    float inv = rsqrtf(vv[i] + 1e-5f);
    float sc = g[i] * inv;
    s[i] = sc;
    t[i] = (fpb[i] - mm[i]) * sc + bb[i];
  }
}
__global__ void k_xt(const float* __restrict__ x, bf_t* __restrict__ xT) {
  long total = (long)Bb * Mm * Cc;
  for (long i = (long)blockIdx.x * blockDim.x + threadIdx.x; i < total; i += (long)gridDim.x * blockDim.x) {
    int c = (int)(i & 255); long bm = i >> 8; int m = (int)(bm & 4095); int b = (int)(bm >> 12);
    xT[i] = f2bf(x[(((long)b * Cc + c) << 12) + m]);
  }
}
__global__ void k_pos(const float* __restrict__ pw, const float* __restrict__ pb, bf_t* __restrict__ combT) {
  long total = (long)Bb * Mm * 128;
  for (long i = (long)blockIdx.x * blockDim.x + threadIdx.x; i < total; i += (long)gridDim.x * blockDim.x) {
    int p = (int)(i & 127); long bm = i >> 7; int m = (int)(bm & 4095); int b = (int)(bm >> 12);
    int hh = m >> 6, ww = m & 63;
    float y  = -1.f + 2.f * (float)hh / 63.f;
    float xx = -1.f + 2.f * (float)ww / 63.f;
    float v = pw[2 * p] * y + pw[2 * p + 1] * xx + pb[p];
    combT[((((long)b) << 12) + m) * 256 + 128 + p] = f2bf(v);
  }
}

// ---------------- comb = [ReLU(BN(fp_w128 . x)) ; pos] in [b][m][ch] ----------------
__global__ __launch_bounds__(256) void k_comb(const bf_t* __restrict__ xT, const bf_t* __restrict__ wfp,
                                              const float* __restrict__ bns, const float* __restrict__ bnt,
                                              bf_t* __restrict__ combT) {
  int wv = threadIdx.x >> 5, lane = threadIdx.x & 31, hf = lane >> 4, lrow = lane & 15;
  int t = blockIdx.x * 8 + wv;                      // 32768 tiles
  int b = t >> 11; int rem = t & 2047; int ot = rem >> 8; int mt = rem & 255;
  int o0 = ot * 16, m0 = mt * 16;
  const bf_t* arow = wfp + (o0 + lrow) * 256;
  const bf_t* bcol = xT + ((((long)b) << 12) + m0 + lrow) * 256;
  f32x8 acc = zero8();
#pragma unroll
  for (int k0 = 0; k0 < 256; k0 += 32)
    acc = wmma_bf(frag_a(arow, k0, hf), frag_b(bcol, k0, hf), acc);
  bf_t tmp[8];
#pragma unroll
  for (int r = 0; r < 8; ++r) {
    int o = o0 + 8 * hf + r;
    float v = acc[r] * bns[o] + bnt[o];
    tmp[r] = f2bf(fmaxf(v, 0.f));
  }
  *(bf16x8*)(combT + ((((long)b) << 12) + m0 + lrow) * 256 + o0 + 8 * hf) = *(bf16x8*)tmp;
}

// ---------------- keysT[b][m][d] = key_w . comb + key_b ----------------
__global__ __launch_bounds__(256) void k_keys(const bf_t* __restrict__ combT, const bf_t* __restrict__ wkey,
                                              const float* __restrict__ keyb, bf_t* __restrict__ keysT) {
  int wv = threadIdx.x >> 5, lane = threadIdx.x & 31, hf = lane >> 4, lrow = lane & 15;
  int t = blockIdx.x * 8 + wv;                      // 16384 tiles
  int b = t >> 10; int rem = t & 1023; int dt = rem >> 8; int mt = rem & 255;
  int d0 = dt * 16, m0 = mt * 16;
  const bf_t* arow = wkey + (d0 + lrow) * 256;
  const bf_t* bcol = combT + ((((long)b) << 12) + m0 + lrow) * 256;
  f32x8 acc = zero8();
#pragma unroll
  for (int k0 = 0; k0 < 256; k0 += 32)
    acc = wmma_bf(frag_a(arow, k0, hf), frag_b(bcol, k0, hf), acc);
  bf_t tmp[8];
#pragma unroll
  for (int r = 0; r < 8; ++r) {
    int d = d0 + 8 * hf + r;
    tmp[r] = f2bf(acc[r] + keyb[d]);
  }
  *(bf16x8*)(keysT + ((((long)b) << 12) + m0 + lrow) * 64 + d0 + 8 * hf) = *(bf16x8*)tmp;
}

// ---------------- vals[b][c][m] = val_w . comb + val_b ----------------
__global__ __launch_bounds__(256) void k_vals(const bf_t* __restrict__ combT, const bf_t* __restrict__ wval,
                                              const float* __restrict__ valb, bf_t* __restrict__ vals) {
  int wv = threadIdx.x >> 5, lane = threadIdx.x & 31, hf = lane >> 4, lrow = lane & 15;
  int t = blockIdx.x * 8 + wv;                      // 65536 tiles
  int b = t >> 12; int rem = t & 4095; int ct = rem >> 8; int mt = rem & 255;
  int c0 = ct * 16, m0 = mt * 16;
  const bf_t* arow = wval + (c0 + lrow) * 256;
  const bf_t* bcol = combT + ((((long)b) << 12) + m0 + lrow) * 256;
  f32x8 acc = zero8();
#pragma unroll
  for (int k0 = 0; k0 < 256; k0 += 32)
    acc = wmma_bf(frag_a(arow, k0, hf), frag_b(bcol, k0, hf), acc);
#pragma unroll
  for (int r = 0; r < 8; ++r) {
    int cch = c0 + r + 8 * hf;
    vals[(((long)b * 256) + cch) * 4096 + m0 + lrow] = f2bf(acc[r] + valb[cch]);
  }
}

// ------- attention: logits(K=16 padded WMMA) -> 3-pass softmax -> head mean -------
__global__ __launch_bounds__(128) void k_attn(const bf_t* __restrict__ qbf, const bf_t* __restrict__ keysT,
                                              float* __restrict__ maps, bf_t* __restrict__ attn) {
  __shared__ float sb[4 * 256];
  int b = blockIdx.x / 19, nt = blockIdx.x % 19, n0 = nt * 16;
  int h = threadIdx.x >> 5, lane = threadIdx.x & 31, hf = lane >> 4, lrow = lane & 15;
  // A = q tile for head h, K padded 16->32 with zeros
  bf16x16 a;
  {
    const bf_t* qp = qbf + (n0 + lrow) * 64 + h * 16;
    a = bzero16();
    if (hf == 0) a = *(const bf16x16*)qp;
  }
  bf16x16 bz = bzero16();
  const bf_t* kbase = keysT + ((((long)b) << 12) + lrow) * 64 + h * 16;
  float rmax[8], rsum[8];
#pragma unroll
  for (int r = 0; r < 8; ++r) { rmax[r] = -3e38f; rsum[r] = 0.f; }
  // pass 1: row max
  for (int mt = 0; mt < 256; ++mt) {
    bf16x16 bv = bz;
    if (hf == 0) bv = *(const bf16x16*)(kbase + (long)mt * 16 * 64);
    f32x8 c = wmma_bf(a, bv, zero8());
#pragma unroll
    for (int r = 0; r < 8; ++r) rmax[r] = fmaxf(rmax[r], c[r]);
  }
#pragma unroll
  for (int r = 0; r < 8; ++r) rmax[r] = redmax16(rmax[r]);
  // pass 2: row sum of exp
  for (int mt = 0; mt < 256; ++mt) {
    bf16x16 bv = bz;
    if (hf == 0) bv = *(const bf16x16*)(kbase + (long)mt * 16 * 64);
    f32x8 c = wmma_bf(a, bv, zero8());
#pragma unroll
    for (int r = 0; r < 8; ++r) rsum[r] += __expf(0.25f * (c[r] - rmax[r]));
  }
#pragma unroll
  for (int r = 0; r < 8; ++r) rsum[r] = 1.f / redsum16(rsum[r]);
  // pass 3: emit, combine heads via LDS
  for (int mt = 0; mt < 256; ++mt) {
    bf16x16 bv = bz;
    if (hf == 0) bv = *(const bf16x16*)(kbase + (long)mt * 16 * 64);
    f32x8 c = wmma_bf(a, bv, zero8());
#pragma unroll
    for (int r = 0; r < 8; ++r)
      sb[h * 256 + (r + 8 * hf) * 16 + lrow] = __expf(0.25f * (c[r] - rmax[r])) * rsum[r];
    __syncthreads();
    for (int t2 = threadIdx.x; t2 < 256; t2 += 128) {
      float v = 0.25f * (sb[t2] + sb[256 + t2] + sb[512 + t2] + sb[768 + t2]);
      int nl = t2 >> 4, ml = t2 & 15, n = n0 + nl, m = mt * 16 + ml;
      if (n < Nn) maps[(((long)b * Nn) + n) * 4096 + m] = v;
      attn[(((long)b * NPAD) + n) * 4096 + m] = (n < Nn) ? f2bf(v) : f2bf(0.f);
    }
    __syncthreads();
  }
}

// ---------------- attention row norms ----------------
__global__ __launch_bounds__(128) void k_norms(const bf_t* __restrict__ attn, float* __restrict__ invn) {
  int idx = blockIdx.x;                              // b*NPAD + n
  const bf_t* p = attn + (long)idx * 4096;
  float s = 0.f;
  for (int i = threadIdx.x; i < 4096; i += 128) { float v = bf2f(p[i]); s += v * v; }
#pragma unroll
  for (int m = 1; m < 32; m <<= 1) s += __shfl_xor(s, m, 32);
  __shared__ float ps[4];
  if ((threadIdx.x & 31) == 0) ps[threadIdx.x >> 5] = s;
  __syncthreads();
  if (threadIdx.x == 0) {
    s = ps[0] + ps[1] + ps[2] + ps[3];
    invn[idx] = 1.f / fmaxf(sqrtf(s), 1e-12f);
  }
}

// ---------------- sf = attn . vals^T ----------------
__global__ __launch_bounds__(256) void k_sf(const bf_t* __restrict__ attn, const bf_t* __restrict__ vals,
                                            bf_t* __restrict__ sfbf, float* __restrict__ osf) {
  int wv = threadIdx.x >> 5, lane = threadIdx.x & 31, hf = lane >> 4, lrow = lane & 15;
  int t = blockIdx.x * 8 + wv;                       // 4864 tiles
  int b = t / 304; int rem = t % 304; int nt = rem / 16; int ct = rem % 16;
  int n0 = nt * 16, c0 = ct * 16;
  const bf_t* arow = attn + (((long)b * NPAD) + n0 + lrow) * 4096;
  const bf_t* bcol = vals + (((long)b * 256) + c0 + lrow) * 4096;
  f32x8 acc = zero8();
#pragma unroll 4
  for (int k0 = 0; k0 < 4096; k0 += 32)
    acc = wmma_bf(frag_a(arow, k0, hf), frag_b(bcol, k0, hf), acc);
  int cc = c0 + lrow;
#pragma unroll
  for (int r = 0; r < 8; ++r) {
    int n = n0 + r + 8 * hf;
    sfbf[(((long)b * NPAD) + n) * 256 + cc] = f2bf(acc[r]);
    if (n < Nn) osf[(((long)b * Nn) + n) * 256 + cc] = acc[r];
  }
}

// ---------------- MLP heads + cls ----------------
__device__ void head_mlp(const bf_t* sfrow, const bf_t* w1, const float* b1,
                         const float* g, const float* bt, const bf_t* w2, const float* b2,
                         float* z1, bf_t* z1b, bf_t* z2b, int wv, int hf, int lrow) {
  // GEMM1: z1[16][256] = sf . w1^T + b1
  for (int ot = wv; ot < 16; ot += 8) {
    int o0 = ot * 16;
    const bf_t* bcol = w1 + (o0 + lrow) * 256;
    f32x8 acc = zero8();
#pragma unroll
    for (int k0 = 0; k0 < 256; k0 += 32)
      acc = wmma_bf(frag_a(sfrow, k0, hf), frag_b(bcol, k0, hf), acc);
    float bias = b1[o0 + lrow];
#pragma unroll
    for (int r = 0; r < 8; ++r) z1[(r + 8 * hf) * 256 + o0 + lrow] = acc[r] + bias;
  }
  __syncthreads();
  // LayerNorm + ReLU -> z1b (bf16)
  {
    int row = threadIdx.x >> 4, l = threadIdx.x & 15;
    float s = 0.f;
#pragma unroll
    for (int j = 0; j < 16; ++j) s += z1[row * 256 + l + 16 * j];
    s = redsum16(s);
    float mu = s * (1.f / 256.f);
    float vs = 0.f;
#pragma unroll
    for (int j = 0; j < 16; ++j) { float d = z1[row * 256 + l + 16 * j] - mu; vs += d * d; }
    vs = redsum16(vs);
    float rstd = rsqrtf(vs * (1.f / 256.f) + 1e-5f);
#pragma unroll
    for (int j = 0; j < 16; ++j) {
      int o = l + 16 * j;
      float v = (z1[row * 256 + o] - mu) * rstd * g[o] + bt[o];
      z1b[row * 256 + o] = f2bf(fmaxf(v, 0.f));
    }
  }
  __syncthreads();
  // GEMM2: z2[16][128] = relu(z1 . w2^T + b2)
  {
    int o0 = wv * 16;
    const bf_t* arow2 = z1b + lrow * 256;
    const bf_t* bcol = w2 + (o0 + lrow) * 256;
    f32x8 acc = zero8();
#pragma unroll
    for (int k0 = 0; k0 < 256; k0 += 32)
      acc = wmma_bf(frag_a(arow2, k0, hf), frag_b(bcol, k0, hf), acc);
    float bias = b2[o0 + lrow];
#pragma unroll
    for (int r = 0; r < 8; ++r)
      z2b[(r + 8 * hf) * 128 + o0 + lrow] = f2bf(fmaxf(acc[r] + bias, 0.f));
  }
  __syncthreads();
}

__global__ __launch_bounds__(256) void k_heads(
    const bf_t* __restrict__ sfbf,
    const bf_t* w1bb, const float* b1bb, const float* gbb, const float* btbb,
    const bf_t* w2bb, const float* b2bb, const float* w3bb, const float* b3bb,
    const bf_t* w1ch, const float* b1ch, const float* gch, const float* btch,
    const bf_t* w2ch, const float* b2ch, const float* w3ch, const float* b3ch,
    const bf_t* wctp, const float* bctp,
    float* __restrict__ oboxes, float* __restrict__ oscores, float* __restrict__ ocls) {
  __shared__ float z1[16 * 256];
  __shared__ bf_t z1b[16 * 256];
  __shared__ bf_t z2b[16 * 128];
  int b = blockIdx.x / 19, nt = blockIdx.x % 19, n0 = nt * 16;
  int wv = threadIdx.x >> 5, lane = threadIdx.x & 31, hf = lane >> 4, lrow = lane & 15;
  const bf_t* sfrow = sfbf + (((long)b * NPAD) + n0 + lrow) * 256;
  // cls = sf . ctp^T + b (waves 0..4 cover 80 outputs)
  if (wv < 5) {
    int o0 = wv * 16;
    const bf_t* bcol = wctp + (o0 + lrow) * 256;
    f32x8 acc = zero8();
#pragma unroll
    for (int k0 = 0; k0 < 256; k0 += 32)
      acc = wmma_bf(frag_a(sfrow, k0, hf), frag_b(bcol, k0, hf), acc);
    float bias = bctp[o0 + lrow];
#pragma unroll
    for (int r = 0; r < 8; ++r) {
      int n = n0 + r + 8 * hf;
      if (n < Nn) ocls[(((long)b * Nn) + n) * 80 + o0 + lrow] = acc[r] + bias;
    }
  }
  __syncthreads();
  // boxes head
  head_mlp(sfrow, w1bb, b1bb, gbb, btbb, w2bb, b2bb, z1, z1b, z2b, wv, hf, lrow);
  if (threadIdx.x < 64) {
    int nl = threadIdx.x >> 2, j = threadIdx.x & 3, n = n0 + nl;
    if (n < Nn) {
      float s = b3bb[j];
      for (int k = 0; k < 128; ++k) s += bf2f(z2b[nl * 128 + k]) * w3bb[j * 128 + k];
      oboxes[(((long)b * Nn) + n) * 4 + j] = s;
    }
  }
  __syncthreads();
  // score head
  head_mlp(sfrow, w1ch, b1ch, gch, btch, w2ch, b2ch, z1, z1b, z2b, wv, hf, lrow);
  if (threadIdx.x < 16) {
    int nl = threadIdx.x, n = n0 + nl;
    if (n < Nn) {
      float s = b3ch[0];
      for (int k = 0; k < 128; ++k) s += bf2f(z2b[nl * 128 + k]) * w3ch[k];
      oscores[((long)b * Nn) + n] = 1.f / (1.f + __expf(-s));
    }
  }
}

// ---------------- diversity: sim = attn.attn^T, scale by invnorms ----------------
__global__ __launch_bounds__(256) void k_sim(const bf_t* __restrict__ attn, const float* __restrict__ invn,
                                             float* __restrict__ divacc) {
  int wv = threadIdx.x >> 5, lane = threadIdx.x & 31, hf = lane >> 4, lrow = lane & 15;
  int t = blockIdx.x * 8 + wv;                       // 5776 tiles exactly (722*8)
  int b = t / 361; int rem = t % 361; int it = rem / 19; int jt = rem % 19;
  int i0 = it * 16, j0 = jt * 16;
  const bf_t* arow = attn + (((long)b * NPAD) + i0 + lrow) * 4096;
  const bf_t* bcol = attn + (((long)b * NPAD) + j0 + lrow) * 4096;
  f32x8 acc = zero8();
#pragma unroll 4
  for (int k0 = 0; k0 < 4096; k0 += 32)
    acc = wmma_bf(frag_a(arow, k0, hf), frag_b(bcol, k0, hf), acc);
  int gj = j0 + lrow;
  float ij = invn[b * NPAD + gj];
  float s = 0.f;
#pragma unroll
  for (int r = 0; r < 8; ++r) {
    int gi = i0 + r + 8 * hf;
    float v = acc[r];
    if (gi == gj) v = 0.f;
    s += v * invn[b * NPAD + gi] * ij;     // padded rows have v == 0 exactly
  }
#pragma unroll
  for (int m = 1; m < 32; m <<= 1) s += __shfl_xor(s, m, 32);
  if (lane == 0) atomicAdd(divacc, s);
}

__global__ void k_div(const float* __restrict__ acc, float* __restrict__ out) {
  if (threadIdx.x == 0) out[0] = acc[0] * (0.1f / (300.f * 299.f));
}

// ---------------- launch ----------------
extern "C" void kernel_launch(void* const* d_in, const int* in_sizes, int n_in,
                              void* d_out, int out_size, void* d_ws, size_t ws_size,
                              hipStream_t stream) {
  const float* x      = (const float*)d_in[0];
  const float* qsrc   = (const float*)d_in[1];
  const float* pos_w  = (const float*)d_in[2];
  const float* pos_b  = (const float*)d_in[3];
  const float* fp_w   = (const float*)d_in[4];
  const float* fp_b   = (const float*)d_in[5];
  const float* bn_g   = (const float*)d_in[6];
  const float* bn_bb  = (const float*)d_in[7];
  const float* bn_m   = (const float*)d_in[8];
  const float* bn_v   = (const float*)d_in[9];
  const float* key_w  = (const float*)d_in[10];
  const float* key_b  = (const float*)d_in[11];
  const float* val_w  = (const float*)d_in[12];
  const float* val_b  = (const float*)d_in[13];
  const float* ctp_w  = (const float*)d_in[14];
  const float* ctp_b  = (const float*)d_in[15];
  const float* bb1_w  = (const float*)d_in[16];
  const float* bb1_b  = (const float*)d_in[17];
  const float* bbln_g = (const float*)d_in[18];
  const float* bbln_b = (const float*)d_in[19];
  const float* bb2_w  = (const float*)d_in[20];
  const float* bb2_b  = (const float*)d_in[21];
  const float* bb3_w  = (const float*)d_in[22];
  const float* bb3_b  = (const float*)d_in[23];
  const float* ch1_w  = (const float*)d_in[24];
  const float* ch1_b  = (const float*)d_in[25];
  const float* chln_g = (const float*)d_in[26];
  const float* chln_b = (const float*)d_in[27];
  const float* ch2_w  = (const float*)d_in[28];
  const float* ch2_b  = (const float*)d_in[29];
  const float* ch3_w  = (const float*)d_in[30];
  const float* ch3_b  = (const float*)d_in[31];

  float* out = (float*)d_out;
  float* o_boxes  = out;                 // 16*300*4
  float* o_scores = out + 19200;         // 16*300
  float* o_cls    = out + 24000;         // 16*300*80
  float* o_maps   = out + 408000;        // 16*300*4096
  float* o_div    = out + 20068800;      // 1
  float* o_sf     = out + 20068801;      // 16*300*256

  char* w = (char*)d_ws; size_t off = 0;
  auto carve = [&](size_t bytes) -> void* {
    void* p = w + off; off = (off + bytes + 255) & ~(size_t)255; return p;
  };
  bf_t* xT    = (bf_t*)carve((size_t)Bb * Mm * Cc * 2);
  bf_t* combT = (bf_t*)carve((size_t)Bb * Mm * HID * 2);
  bf_t* keysT = (bf_t*)carve((size_t)Bb * Mm * SDd * 2);
  bf_t* vals  = (bf_t*)carve((size_t)Bb * HID * Mm * 2);
  bf_t* attn  = (bf_t*)carve((size_t)Bb * NPAD * Mm * 2);
  bf_t* sfbf  = (bf_t*)carve((size_t)Bb * NPAD * HID * 2);
  float* invn = (float*)carve((size_t)Bb * NPAD * 4);
  bf_t* wfp   = (bf_t*)carve((size_t)128 * 256 * 2);
  bf_t* wkey  = (bf_t*)carve((size_t)64 * 256 * 2);
  bf_t* wval  = (bf_t*)carve((size_t)256 * 256 * 2);
  bf_t* qbf   = (bf_t*)carve((size_t)NPAD * 64 * 2);
  bf_t* wbb1  = (bf_t*)carve((size_t)256 * 256 * 2);
  bf_t* wch1  = (bf_t*)carve((size_t)256 * 256 * 2);
  bf_t* wbb2  = (bf_t*)carve((size_t)128 * 256 * 2);
  bf_t* wch2  = (bf_t*)carve((size_t)128 * 256 * 2);
  bf_t* wctp  = (bf_t*)carve((size_t)80 * 256 * 2);
  float* bns  = (float*)carve(128 * 4);
  float* bnt  = (float*)carve(128 * 4);
  float* dacc = (float*)carve(256);
  (void)ws_size; (void)n_in; (void)in_sizes; (void)out_size;

  // weight conversions
  k_cast<<<128, 256, 0, stream>>>(fp_w, wfp, 128 * 256);   // only first 128 rows used
  k_cast<<<64, 256, 0, stream>>>(key_w, wkey, 64 * 256);
  k_cast<<<256, 256, 0, stream>>>(val_w, wval, 256 * 256);
  k_cast<<<256, 256, 0, stream>>>(bb1_w, wbb1, 256 * 256);
  k_cast<<<256, 256, 0, stream>>>(ch1_w, wch1, 256 * 256);
  k_cast<<<128, 256, 0, stream>>>(bb2_w, wbb2, 128 * 256);
  k_cast<<<128, 256, 0, stream>>>(ch2_w, wch2, 128 * 256);
  k_cast<<<80, 256, 0, stream>>>(ctp_w, wctp, 80 * 256);
  k_qpad<<<(NPAD * 64 + 255) / 256, 256, 0, stream>>>(qsrc, qbf);
  k_bnfold<<<1, 128, 0, stream>>>(bn_g, bn_bb, bn_m, bn_v, fp_b, bns, bnt, dacc);

  // data staging
  k_xt<<<8192, 256, 0, stream>>>(x, xT);
  k_pos<<<8192, 256, 0, stream>>>(pos_w, pos_b, combT);

  // big GEMM pipeline
  k_comb<<<4096, 256, 0, stream>>>(xT, wfp, bns, bnt, combT);
  k_keys<<<2048, 256, 0, stream>>>(combT, wkey, key_b, keysT);
  k_vals<<<8192, 256, 0, stream>>>(combT, wval, val_b, vals);
  k_attn<<<Bb * 19, 128, 0, stream>>>(qbf, keysT, o_maps, attn);
  k_norms<<<Bb * NPAD, 128, 0, stream>>>(attn, invn);
  k_sf<<<608, 256, 0, stream>>>(attn, vals, sfbf, o_sf);
  k_heads<<<Bb * 19, 256, 0, stream>>>(sfbf,
      wbb1, bb1_b, bbln_g, bbln_b, wbb2, bb2_b, bb3_w, bb3_b,
      wch1, ch1_b, chln_g, chln_b, wch2, ch2_b, ch3_w, ch3_b,
      wctp, ctp_b, o_boxes, o_scores, o_cls);
  k_sim<<<722, 256, 0, stream>>>(attn, invn, dacc);
  k_div<<<1, 64, 0, stream>>>(dacc, o_div);
}
